// KWS_LSTM_25941602468232
// MI455X (gfx1250) — compile-verified, matrix operand order checked by reading, not run on hardware
//
#include <hip/hip_runtime.h>

// ---------------- problem constants ----------------
constexpr int Dn = 80, Hn = 256, Tn = 200, Bn = 512, Ln = 8, Gn = 1024;
constexpr int BM = 64;          // batch rows per workgroup
constexpr int BN = 64;          // hidden units per workgroup
constexpr int NCOL = 4 * BN;    // gate columns per workgroup = 256
constexpr int KX = 96;          // D padded to multiple of 32
constexpr int KH = 256;         // hidden K
constexpr int KTOT = KX + KH;   // 352, 11 chunks of 32
constexpr int NKC = KTOT / 32;  // 11

// ---------------- LDS layout (dynamic) ----------------
constexpr unsigned WLDS_BYTES  = NCOL * KTOT * 2;  // 180224
constexpr unsigned XBUF_BYTES  = BM * KX * 2;      // 12288
constexpr unsigned HBUF_BYTES  = BM * KH * 2;      // 32768
constexpr unsigned GATES_BYTES = BM * NCOL * 4;    // 65536
constexpr unsigned SMEM_BYTES  = WLDS_BYTES + XBUF_BYTES + HBUF_BYTES + GATES_BYTES; // 290816 < 320K

// ---------------- workspace layout (bytes) ----------------
constexpr size_t MAXES_OFF = 0;                              // 32 f32
constexpr size_t BAR_OFF   = 256;                            // 64 groups * 2 u32
constexpr size_t BIAS_OFF  = 1024;                           // L*G f32
constexpr size_t WQ_OFF    = BIAS_OFF + (size_t)Ln * Gn * 4;            // 33792
constexpr size_t XQ_OFF    = WQ_OFF + (size_t)Ln * 4 * Hn * KTOT * 2;   // +5767168
constexpr size_t HQ_OFF    = XQ_OFF + (size_t)Tn * Bn * KX * 2;         // +19660800
constexpr size_t HT_OFF    = HQ_OFF + (size_t)2 * Ln * Bn * KH * 2;     // +4194304

// ---------------- WMMA vector types ----------------
typedef __attribute__((ext_vector_type(16))) __bf16 v16bf;
typedef __attribute__((ext_vector_type(8)))  __bf16 v8bf;
typedef __attribute__((ext_vector_type(8)))  float  v8f;

// ---------------- CDNA5 async global->LDS copy ----------------
__device__ inline void async_g2l_b128(void* ldsp, const void* g) {
  unsigned off = (unsigned)(size_t)ldsp;  // low 32 bits of LDS flat addr = LDS offset
  asm volatile("global_load_async_to_lds_b128 %0, %1, off"
               :: "v"(off), "v"(g) : "memory");
}
#define ASYNC_WAIT() asm volatile("s_wait_asynccnt 0" ::: "memory")

// ---------------- math helpers ----------------
__device__ inline float sigm(float x) { return 1.0f / (1.0f + __expf(-x)); }
__device__ inline float tanh_fast(float x) {
  x = fminf(fmaxf(x, -15.0f), 15.0f);
  float e = __expf(2.0f * x);
  return (e - 1.0f) / (e + 1.0f);
}
// fake-quant with PACT clip: all bit-widths here are 8 -> step = 128
__device__ inline float qq(float x, float a) {
  const float s = 128.0f;
  float y = x / a;
  y = fminf(fmaxf(y, -1.0f + 1.0f / s), 1.0f - 1.0f / s);
  return rintf(y * s) * (a / s);
}
// weight fake-quant, xr = 1.0
__device__ inline float qw(float w) {
  float y = fminf(fmaxf(w, -1.0f + 1.0f / 128.0f), 1.0f - 1.0f / 128.0f);
  return rintf(y * 128.0f) * (1.0f / 128.0f);
}
// deterministic hash RNG -> approx N(0,1) (Irwin-Hall of 4 uniforms)
__device__ inline unsigned hmix(unsigned x) {
  x ^= x >> 16; x *= 0x7feb352du; x ^= x >> 15; x *= 0x846ca68bu; x ^= x >> 16; return x;
}
__device__ inline float nrand(unsigned a, unsigned b) {
  unsigned h1 = hmix(a * 0x9e3779b9u ^ (b + 0x85ebca6bu));
  unsigned h2 = hmix(h1 ^ 0xc2b2ae35u);
  unsigned h3 = hmix(h2 + 0x165667b1u);
  unsigned h4 = hmix(h1 + 0x27d4eb2fu);
  const float k = 1.0f / 16777216.0f;
  return ((h1 >> 8) * k + (h2 >> 8) * k + (h3 >> 8) * k + (h4 >> 8) * k - 2.0f) * 1.7320508f;
}

// ---------------- fragment load from LDS (ISA 16-bit A/B layout) ----------------
// element e of lane L maps to K = e + (e<8?0:8) + (L>=16?8:0)  => two contiguous 8-elem runs
__device__ inline v16bf load_frag(const __bf16* base, int stride_elems, int row0, int kbase, int lane) {
  const __bf16* p = base + (size_t)(row0 + (lane & 15)) * stride_elems + kbase + ((lane >> 4) << 3);
  union { v16bf v; v8bf h[2]; } u;
  u.h[0] = *reinterpret_cast<const v8bf*>(p);        // K runs {0..7}/{8..15}
  u.h[1] = *reinterpret_cast<const v8bf*>(p + 16);   // K runs {16..23}/{24..31}
  return u.v;
}

// ---------------- group barrier (4 WGs sharing a batch tile of one layer) ----------------
__device__ inline void group_barrier(unsigned* bars, int group, unsigned target) {
  __syncthreads();
  // CDNA5 cluster path: NOP when launched unclustered (ClusterID==0)
  if (__builtin_amdgcn_cluster_id_x() != 0) __builtin_amdgcn_s_cluster_barrier();
  if (threadIdx.x == 0) {
    unsigned* cnt   = bars + group * 2;
    unsigned* epoch = bars + group * 2 + 1;
    unsigned prev = __hip_atomic_fetch_add(cnt, 1u, __ATOMIC_ACQ_REL, __HIP_MEMORY_SCOPE_AGENT);
    if (prev == 3u) {
      __hip_atomic_store(cnt, 0u, __ATOMIC_RELAXED, __HIP_MEMORY_SCOPE_AGENT);
      __hip_atomic_fetch_add(epoch, 1u, __ATOMIC_ACQ_REL, __HIP_MEMORY_SCOPE_AGENT);
    }
    while (__hip_atomic_load(epoch, __ATOMIC_ACQUIRE, __HIP_MEMORY_SCOPE_AGENT) < target) {
      __builtin_amdgcn_s_sleep(2);
    }
  }
  __syncthreads();
}

// ================= prep kernels =================
__global__ void reduce_max_kernel(const float* __restrict__ Wih, const float* __restrict__ Whh,
                                  const float* __restrict__ bih, const float* __restrict__ bhh,
                                  float* __restrict__ maxes) {
  __shared__ float red[256];
  int l = blockIdx.x >> 2, which = blockIdx.x & 3, tid = threadIdx.x;
  const float* p; int n;
  if (which == 0)      { p = Wih + (size_t)l * Dn * Gn; n = Dn * Gn; }
  else if (which == 1) { p = Whh + (size_t)l * Hn * Gn; n = Hn * Gn; }
  else if (which == 2) { p = bih + (size_t)l * Gn;      n = Gn; }
  else                 { p = bhh + (size_t)l * Gn;      n = Gn; }
  float m = -3.0e38f;
  for (int i = tid; i < n; i += 256) m = fmaxf(m, p[i]);
  red[tid] = m; __syncthreads();
  for (int s = 128; s > 0; s >>= 1) { if (tid < s) red[tid] = fmaxf(red[tid], red[tid + s]); __syncthreads(); }
  if (tid == 0) maxes[blockIdx.x] = red[0];
}

// Wq layout: [L][4 gates][H][KTOT] bf16; K 0..79 = W_ih rows, 80..95 = zero pad, 96..351 = W_hh rows
__global__ void prep_w_kernel(const float* __restrict__ Wih, const float* __restrict__ Whh,
                              const float* __restrict__ maxes, __bf16* __restrict__ Wq) {
  const size_t total = (size_t)Ln * 4 * Hn * KTOT;
  for (size_t idx = (size_t)blockIdx.x * 256 + threadIdx.x; idx < total; idx += (size_t)gridDim.x * 256) {
    int k = (int)(idx % KTOT);
    size_t r = idx / KTOT;
    int n = (int)(r % Hn); r /= Hn;
    int q = (int)(r % 4);
    int l = (int)(r / 4);
    int g = q * Hn + n;
    float val = 0.0f;
    if (k < Dn) {
      float w = Wih[((size_t)l * Dn + k) * Gn + g];
      val = qw(w) + nrand(0x1234u + (unsigned)l * 2u, (unsigned)idx) * maxes[l * 4 + 0] * 0.1f;
    } else if (k >= KX) {
      float w = Whh[((size_t)l * Hn + (k - KX)) * Gn + g];
      val = qw(w) + nrand(0x1235u + (unsigned)l * 2u, (unsigned)idx) * maxes[l * 4 + 1] * 0.1f;
    }
    Wq[idx] = (__bf16)val;
  }
}

__global__ void prep_bias_kernel(const float* __restrict__ bih, const float* __restrict__ bhh,
                                 const float* __restrict__ maxes, float* __restrict__ biasq) {
  int idx = blockIdx.x * 256 + threadIdx.x;
  if (idx < Ln * Gn) {
    int l = idx / Gn;
    biasq[idx] = qw(bih[idx]) + qw(bhh[idx]) +
                 (nrand(0x777u, (unsigned)idx) * maxes[l * 4 + 2] +
                  nrand(0x778u, (unsigned)idx) * maxes[l * 4 + 3]) * 0.1f;
  }
}

// Xq: [T][B][KX] bf16, input PACT(8-bit, a0)-quantized, zero-padded K 80..95
__global__ void prep_x_kernel(const float* __restrict__ x, const float* __restrict__ a_lstm,
                              __bf16* __restrict__ Xq) {
  float a0 = a_lstm[0];
  const size_t total = (size_t)Tn * Bn * KX;
  for (size_t idx = (size_t)blockIdx.x * 256 + threadIdx.x; idx < total; idx += (size_t)gridDim.x * 256) {
    int k = (int)(idx % KX);
    size_t tb = idx / KX;
    float v = 0.0f;
    if (k < Dn) v = qq(x[tb * Dn + k], a0);
    Xq[idx] = (__bf16)v;
  }
}

// ================= main persistent LSTM kernel =================
// grid: 256 blocks = L(8) x batch-tiles(8) x hidden-tiles(4); block = 256 threads (8 wave32)
__global__ __launch_bounds__(256, 1) void lstm_kernel(
    const __bf16* __restrict__ Xq, const __bf16* __restrict__ Wq,
    const float* __restrict__ biasq, const float* __restrict__ a_lstm,
    __bf16* __restrict__ Hq, float* __restrict__ hT_out, unsigned* bars) {
  extern __shared__ __align__(16) char smem[];
  __bf16* Wlds  = (__bf16*)smem;                                        // [NCOL][KTOT]
  __bf16* xbuf  = (__bf16*)(smem + WLDS_BYTES);                         // [BM][KX]
  __bf16* hbuf  = (__bf16*)(smem + WLDS_BYTES + XBUF_BYTES);            // [BM][KH]
  float*  gates = (float*)(smem + WLDS_BYTES + XBUF_BYTES + HBUF_BYTES);// [BM][NCOL]

  const int tid  = threadIdx.x;
  const int lane = tid & 31;
  const int wave = tid >> 5;
  const int l  = blockIdx.x >> 5;
  const int bm = (blockIdx.x >> 2) & 7;
  const int hn = blockIdx.x & 3;
  const int group = (l << 3) | bm;

  // ---- one-time: async-copy this WG's weight slice into LDS ----
  {
    const char* gW = (const char*)Wq;
    for (int idx = tid; idx < NCOL * (KTOT / 8); idx += 256) {
      int c  = idx / (KTOT / 8);           // local gate col 0..255
      int ch = idx % (KTOT / 8);           // 16-byte chunk along K
      int q = c >> 6, j = c & 63;
      size_t gcol = ((size_t)(l * 4 + q)) * Hn + hn * 64 + j;
      async_g2l_b128(Wlds + (size_t)c * KTOT + ch * 8, gW + (gcol * KTOT + (size_t)ch * 8) * 2);
    }
  }
  ASYNC_WAIT();
  __syncthreads();

  const float* a = a_lstm + l * 11;
  const float a2 = a[2], a3 = a[3], a4 = a[4], a5 = a[5], a6 = a[6];
  const float a7 = a[7], a8 = a[8], a9 = a[9], a10 = a[10];

  float creg[16];
#pragma unroll
  for (int i = 0; i < 16; ++i) creg[i] = 0.0f;

  const int b_el = tid & 63;   // elementwise: batch row
  const int nblk = tid >> 6;   // elementwise: hidden sub-block (16 units each)
  const int mtile = wave & 3;  // GEMM: 16-row batch tile
  const int nhalf = wave >> 2; // GEMM: 128-col half

  for (int t = 0; t < Tn; ++t) {
    group_barrier(bars, group, (unsigned)(t + 1));

    const __bf16* Hread = Hq + (size_t)(t & 1) * Ln * Bn * KH;
    __bf16* Hwrite      = Hq + (size_t)((t + 1) & 1) * Ln * Bn * KH;

    // async load x tile (contiguous 12KB) and h tile (contiguous 32KB)
    const char* gx = (const char*)(Xq + ((size_t)t * Bn + bm * BM) * KX);
    for (int i = tid; i < (BM * KX * 2) / 16; i += 256)
      async_g2l_b128((char*)xbuf + i * 16, gx + i * 16);
    const char* gh = (const char*)(Hread + ((size_t)l * Bn + bm * BM) * KH);
    for (int i = tid; i < (BM * KH * 2) / 16; i += 256)
      async_g2l_b128((char*)hbuf + i * 16, gh + i * 16);
    if (t + 1 < Tn)
      __builtin_prefetch((const char*)(Xq + ((size_t)(t + 1) * Bn + bm * BM) * KX) + tid * 64, 0, 1);
    ASYNC_WAIT();
    __syncthreads();

    // ---- WMMA GEMM: gates[64 x 256] += [x|h] (64 x 352) @ W (352 x 256) ----
    v8f acc[8];
#pragma unroll
    for (int nt = 0; nt < 8; ++nt)
#pragma unroll
      for (int r = 0; r < 8; ++r) acc[nt][r] = 0.0f;

    for (int kc = 0; kc < NKC; ++kc) {
      v16bf afrag = (kc < 3)
          ? load_frag(xbuf, KX, mtile * 16, kc * 32, lane)
          : load_frag(hbuf, KH, mtile * 16, (kc - 3) * 32, lane);
#pragma unroll
      for (int nt = 0; nt < 8; ++nt) {
        v16bf bfrag = load_frag(Wlds, KTOT, (nhalf * 8 + nt) * 16, kc * 32, lane);
        acc[nt] = __builtin_amdgcn_wmma_f32_16x16x32_bf16(
            false, afrag, false, bfrag, (short)0, acc[nt], false, false);
      }
    }

    // dump accumulators (ISA C-layout: VGPR r -> M = r + 8*(lane>=16), N = lane&15)
#pragma unroll
    for (int nt = 0; nt < 8; ++nt) {
      int col = (nhalf * 8 + nt) * 16 + (lane & 15);
      int rbase = mtile * 16 + ((lane >> 4) << 3);
#pragma unroll
      for (int r = 0; r < 8; ++r)
        gates[(rbase + r) * NCOL + col] = acc[nt][r];
    }
    __syncthreads();

    // ---- elementwise LSTM cell (gate col order: i, j, f, o) ----
    const float* bias = biasq + (size_t)l * Gn;
#pragma unroll 4
    for (int i = 0; i < 16; ++i) {
      int n = nblk * 16 + i;
      float gi = gates[b_el * NCOL + n]       + bias[0 * Hn + hn * 64 + n];
      float gj = gates[b_el * NCOL + 64 + n]  + bias[1 * Hn + hn * 64 + n];
      float gf = gates[b_el * NCOL + 128 + n] + bias[2 * Hn + hn * 64 + n];
      float go = gates[b_el * NCOL + 192 + n] + bias[3 * Hn + hn * 64 + n];
      float fg = qq(sigm(gf), a2);
      float ig = qq(sigm(gi), a3);
      float jg = qq(tanh_fast(gj), a4);
      float og = qq(sigm(go), a5);
      float gc = qq(creg[i] * fg, a6);
      float ai = qq(ig * jg, a7);
      float nc = qq(gc + ai, a8);
      float ac = qq(tanh_fast(nc), a9);
      float nh = qq(ac * og, a10);
      creg[i] = nc;
      Hwrite[((size_t)l * Bn + bm * BM + b_el) * KH + hn * 64 + n] = (__bf16)nh;
      if (t == Tn - 1)
        hT_out[((size_t)l * Bn + bm * BM + b_el) * Hn + hn * 64 + n] = nh;
    }
    __threadfence();  // release h stores before next barrier arrival
  }
}

// ================= FC heads =================
__global__ void fc_kernel(const float* __restrict__ hT,
                          const float* __restrict__ fcW1, const float* __restrict__ fcb1,
                          const float* __restrict__ fcW2, const float* __restrict__ fcb2,
                          const float* __restrict__ fc_a, float* __restrict__ out) {
  __shared__ float red[256];
  int l = blockIdx.x, tid = threadIdx.x;
  int nc = (l < 4) ? 1 : 2;
  const float* W  = (l < 4) ? fcW1 + (size_t)l * Hn : fcW2 + (size_t)(l - 4) * Hn * 2;
  const float* bb = (l < 4) ? fcb1 + l : fcb2 + (l - 4) * 2;
  float a1 = fc_a[l * 2 + 0], a2 = fc_a[l * 2 + 1];
  float m = -3.0e38f;
  for (int i = tid; i < Hn * nc; i += 256) m = fmaxf(m, W[i]);
  red[tid] = m; __syncthreads();
  for (int s = 128; s > 0; s >>= 1) { if (tid < s) red[tid] = fmaxf(red[tid], red[tid + s]); __syncthreads(); }
  float wmax = red[0];
  float bmax = (nc == 2) ? fmaxf(bb[0], bb[1]) : bb[0];
  for (int b = tid; b < Bn; b += 256) {
    for (int j = 0; j < nc; ++j) {
      float acc = 0.0f;
      for (int k = 0; k < Hn; ++k) {
        float xq = qq(hT[((size_t)l * Bn + b) * Hn + k], a1);
        float w  = qw(W[k * nc + j]) + nrand(0x5135u + (unsigned)l, (unsigned)(k * nc + j)) * wmax * 0.1f;
        acc += xq * w;
      }
      float beff = qw(bb[j]) + nrand(0xb1a5u + (unsigned)l, (unsigned)j) * bmax * 0.1f;
      int col = (l < 4) ? l : 4 + (l - 4) * 2 + j;
      out[b * 12 + col] = qq(acc + beff, a2);
    }
  }
}

// ================= launcher =================
extern "C" void kernel_launch(void* const* d_in, const int* in_sizes, int n_in,
                              void* d_out, int out_size, void* d_ws, size_t ws_size,
                              hipStream_t stream) {
  const float* inputs = (const float*)d_in[0];
  const float* W_ih   = (const float*)d_in[1];
  const float* W_hh   = (const float*)d_in[2];
  const float* b_ih   = (const float*)d_in[3];
  const float* b_hh   = (const float*)d_in[4];
  const float* a_lstm = (const float*)d_in[5];
  const float* fcW1   = (const float*)d_in[6];
  const float* fcb1   = (const float*)d_in[7];
  const float* fcW2   = (const float*)d_in[8];
  const float* fcb2   = (const float*)d_in[9];
  const float* fc_a   = (const float*)d_in[10];

  char* ws = (char*)d_ws;
  float*    maxes = (float*)(ws + MAXES_OFF);
  unsigned* bars  = (unsigned*)(ws + BAR_OFF);
  float*    biasq = (float*)(ws + BIAS_OFF);
  __bf16*   Wq    = (__bf16*)(ws + WQ_OFF);
  __bf16*   Xq    = (__bf16*)(ws + XQ_OFF);
  __bf16*   Hq    = (__bf16*)(ws + HQ_OFF);
  float*    hT    = (float*)(ws + HT_OFF);
  float*    out   = (float*)d_out;

  hipMemsetAsync(bars, 0, 64 * 2 * sizeof(unsigned), stream);
  hipMemsetAsync(Hq, 0, (size_t)Ln * Bn * KH * 2, stream);  // parity-0 h buffer = h_{-1} = 0

  hipFuncSetAttribute(reinterpret_cast<const void*>(lstm_kernel),
                      hipFuncAttributeMaxDynamicSharedMemorySize, (int)SMEM_BYTES);

  reduce_max_kernel<<<32, 256, 0, stream>>>(W_ih, W_hh, b_ih, b_hh, maxes);
  prep_w_kernel<<<2048, 256, 0, stream>>>(W_ih, W_hh, maxes, Wq);
  prep_bias_kernel<<<32, 256, 0, stream>>>(b_ih, b_hh, maxes, biasq);
  prep_x_kernel<<<4096, 256, 0, stream>>>(inputs, a_lstm, Xq);
  lstm_kernel<<<256, 256, SMEM_BYTES, stream>>>(Xq, Wq, biasq, a_lstm, Hq, hT, bars);
  fc_kernel<<<8, 256, 0, stream>>>(hT, fcW1, fcb1, fcW2, fcb2, fc_a, out);
}
